// GNNAnalyzerModel_62526133895606
// MI455X (gfx1250) — matrix operand order repaired and checked by reading, not compiled, not run on hardware
//
#include <hip/hip_runtime.h>
#include <hip/hip_bf16.h>

// ---------------------------------------------------------------------------
// Types for WMMA fragments (wave32, gfx1250)
// ---------------------------------------------------------------------------
typedef __attribute__((ext_vector_type(16))) _Float16 v16h;
typedef __attribute__((ext_vector_type(8)))  _Float16 v8h;
typedef __attribute__((ext_vector_type(8)))  float    v8f;

#define BM 128
#define BN 128
#define BK 32

__device__ __forceinline__ float wave_sum32(float v) {
    #pragma unroll
    for (int off = 16; off > 0; off >>= 1) v += __shfl_xor(v, off, 32);
    return v;
}

// monotone encoding of f32 for unsigned atomicMax
__device__ __forceinline__ unsigned enc_f32(float x) {
    unsigned u = __float_as_uint(x);
    return (u & 0x80000000u) ? ~u : (u | 0x80000000u);
}
__device__ __forceinline__ float dec_f32(unsigned u) {
    u = (u & 0x80000000u) ? (u & 0x7FFFFFFFu) : ~u;
    return __uint_as_float(u);
}

// ---------------------------------------------------------------------------
// f32 -> f16 cast
// ---------------------------------------------------------------------------
__global__ void k_cast_h(const float* __restrict__ in, _Float16* __restrict__ out, int n) {
    int t = blockIdx.x * blockDim.x + threadIdx.x;
    if (t < n) out[t] = (_Float16)in[t];
}

// ---------------------------------------------------------------------------
// embedding gather: xf/xh[n, c] = emb[node_ids[n], c]
// ---------------------------------------------------------------------------
__global__ void k_embed(const int* __restrict__ ids, const float* __restrict__ emb,
                        float* __restrict__ xf, _Float16* __restrict__ xh,
                        int nodes, int Dd) {
    int t = blockIdx.x * blockDim.x + threadIdx.x;
    if (t >= nodes * Dd) return;
    int n = t / Dd, c = t - n * Dd;
    float v = emb[(size_t)ids[n] * Dd + c];
    xf[t] = v;
    xh[t] = (_Float16)v;
}

// ---------------------------------------------------------------------------
// WMMA GEMM: C[M,Nc] = A[M,K] * B[K,Nc]
//   A,B f16 row-major, C f32 row-major. M%128==0, Nc%128==0, K%32==0.
//   256 threads = 8 waves; wave w owns rows [w*16,w*16+16) x all 128 cols of
//   the block tile -> 8 x v_wmma_f32_16x16x32_f16 per K-step.
//   A fragments: direct global loads (no cross-wave reuse -> no LDS staging).
//   B tile: shared by all 8 waves -> double-buffered transposed LDS staging;
//   one barrier per K-step, next tile staged while current is consumed.
// ---------------------------------------------------------------------------
__global__ __launch_bounds__(256) void k_gemm_wmma(
    const _Float16* __restrict__ A, const _Float16* __restrict__ B,
    float* __restrict__ C, int M, int Nc, int K) {
    __shared__ __align__(16) _Float16 BsT[2][BN * BK];  // [n][k] transposed tile

    const int tid  = threadIdx.x;
    const int lane = tid & 31;
    const int wave = tid >> 5;          // 0..7
    const int m0   = blockIdx.x * BM;
    const int n0   = blockIdx.y * BN;

    v8f acc[8] = {};

    // B staging: thread t loads B[k0+brow, n0+bcol .. +16) (32B, coalesced)
    const int brow = tid >> 3;          // 0..31
    const int bcol = (tid & 7) * 16;    // 0,16,...,112

    // fragment addressing (ISA 7.12.2 layouts)
    const int row_a  = m0 + wave * 16 + (lane & 15);
    const int base_a = (lane < 16) ? 0 : 8;    // A: K groups {0..7,16..23} / {8..15,24..31}
    const int kb     = (lane < 16) ? 0 : 8;    // B: K groups {0..7,16..23} / {8..15,24..31}
    const _Float16* Arow = A + (size_t)row_a * K;

    auto stage = [&](int k0, int buf) {
        const _Float16* src = B + (size_t)(k0 + brow) * Nc + (n0 + bcol);
        v8h b0 = *(const v8h*)src;
        v8h b1 = *(const v8h*)(src + 8);
        _Float16* dst = &BsT[buf][0];
        #pragma unroll
        for (int j = 0; j < 8; ++j) dst[(bcol + j) * BK + brow] = b0[j];
        #pragma unroll
        for (int j = 0; j < 8; ++j) dst[(bcol + 8 + j) * BK + brow] = b1[j];
    };

    stage(0, 0);
    int p = 0;
    for (int k0 = 0; k0 < K; k0 += BK, p ^= 1) {
        __syncthreads();                       // staged BsT[p] visible to all waves
        if (k0 + BK < K) {
            stage(k0 + BK, p ^ 1);             // overlap next-tile staging with WMMAs
            if (k0 + 2 * BK < K)               // global_prefetch_b8 two tiles ahead
                __builtin_prefetch(B + (size_t)(k0 + 2 * BK + brow) * Nc + (n0 + bcol), 0, 0);
        }

        // A fragment 16x32 straight from global (two b128 loads per lane)
        v8h alo = *(const v8h*)(Arow + k0 + base_a);
        v8h ahi = *(const v8h*)(Arow + k0 + base_a + 16);
        v16h af = __builtin_shufflevector(alo, ahi, 0,1,2,3,4,5,6,7,8,9,10,11,12,13,14,15);

        const _Float16* bb = &BsT[p][0];
        #pragma unroll
        for (int n = 0; n < 8; ++n) {
            const int col = n * 16 + (lane & 15);
            v8h blo = *(const v8h*)&bb[col * BK + kb];
            v8h bhi = *(const v8h*)&bb[col * BK + kb + 16];
            v16h bf = __builtin_shufflevector(blo, bhi, 0,1,2,3,4,5,6,7,8,9,10,11,12,13,14,15);
            acc[n] = __builtin_amdgcn_wmma_f32_16x16x32_f16(
                false, af, false, bf, (short)0, acc[n], false, false);
        }
    }

    // write D (16x16 f32 layout: VGPR r -> M = r + (lane<16?0:8), N = lane&15)
    const int rbase = m0 + wave * 16 + ((lane < 16) ? 0 : 8);
    #pragma unroll
    for (int n = 0; n < 8; ++n) {
        const int col = n0 + n * 16 + (lane & 15);
        #pragma unroll
        for (int r = 0; r < 8; ++r)
            C[(size_t)(rbase + r) * Nc + col] = acc[n][r];
    }
}

// ---------------------------------------------------------------------------
// per-(node, head) attention scores: es = <h, a_src>, ed = <h, a_dst>
// one wave per (node, head)
// ---------------------------------------------------------------------------
__global__ void k_scores(const float* __restrict__ h, const float* __restrict__ a_s,
                         const float* __restrict__ a_d, float* __restrict__ es,
                         float* __restrict__ ed, int nodes, int heads, int ch) {
    int gt = blockIdx.x * blockDim.x + threadIdx.x;
    int w = gt >> 5, lane = gt & 31;
    if (w >= nodes * heads) return;
    int nd = w / heads, hd = w - nd * heads;
    const float* hp  = h + (size_t)nd * (heads * ch) + hd * ch;
    const float* asp = a_s + hd * ch;
    const float* adp = a_d + hd * ch;
    float s = 0.f, d = 0.f;
    for (int c = lane; c < ch; c += 32) {
        float v = hp[c];
        s += v * asp[c];
        d += v * adp[c];
    }
    s = wave_sum32(s);
    d = wave_sum32(d);
    if (lane == 0) { es[w] = s; ed[w] = d; }
}

// ---------------------------------------------------------------------------
// edge pass 1: segment max of leaky_relu(es[src]+ed[dst]) over dst
// ---------------------------------------------------------------------------
__global__ void k_edge_max(const int* __restrict__ ei, const float* __restrict__ es,
                           const float* __restrict__ ed, unsigned* __restrict__ maxd,
                           int E, int Etot, int heads) {
    int t = blockIdx.x * blockDim.x + threadIdx.x;
    if (t >= Etot * heads) return;
    int e = t / heads, hd = t - e * heads;
    int s, d;
    if (e < E) { s = ei[e]; d = ei[E + e]; } else { s = d = e - E; }
    float v = es[s * heads + hd] + ed[d * heads + hd];
    v = (v > 0.f) ? v : 0.2f * v;
    atomicMax(maxd + d * heads + hd, enc_f32(v));
}

// ---------------------------------------------------------------------------
// edge pass 2: p = exp(e - max[dst]); store p, segment-sum over dst
// ---------------------------------------------------------------------------
__global__ void k_edge_p(const int* __restrict__ ei, const float* __restrict__ es,
                         const float* __restrict__ ed, const unsigned* __restrict__ maxd,
                         float* __restrict__ pe, float* __restrict__ sumd,
                         int E, int Etot, int heads) {
    int t = blockIdx.x * blockDim.x + threadIdx.x;
    if (t >= Etot * heads) return;
    int e = t / heads, hd = t - e * heads;
    int s, d;
    if (e < E) { s = ei[e]; d = ei[E + e]; } else { s = d = e - E; }
    float v = es[s * heads + hd] + ed[d * heads + hd];
    v = (v > 0.f) ? v : 0.2f * v;
    float m = dec_f32(maxd[d * heads + hd]);
    float p = __expf(v - m);
    pe[t] = p;
    atomicAdd(sumd + d * heads + hd, p);
}

// ---------------------------------------------------------------------------
// aggregation: agg[dst, c] += h[src, c] * alpha(edge, head(c)); wave per edge
// ---------------------------------------------------------------------------
__global__ void k_aggregate(const int* __restrict__ ei, const float* __restrict__ h,
                            const float* __restrict__ pe, const float* __restrict__ sumd,
                            float* __restrict__ agg, int E, int Etot, int heads, int ch) {
    int gt = blockIdx.x * blockDim.x + threadIdx.x;
    int w = gt >> 5, lane = gt & 31;
    if (w >= Etot) return;
    int s, d;
    if (w < E) { s = ei[w]; d = ei[E + w]; } else { s = d = w - E; }
    const int Dd = heads * ch;
    for (int c = lane; c < Dd; c += 32) {
        int hd = c / ch;
        float alpha = pe[(size_t)w * heads + hd] / (sumd[d * heads + hd] + 1e-16f);
        atomicAdd(agg + (size_t)d * Dd + c, h[(size_t)s * Dd + c] * alpha);
    }
}

// ---------------------------------------------------------------------------
// bias + relu (+ residual), write f32 and f16 views of the new features
// ---------------------------------------------------------------------------
__global__ void k_post(const float* __restrict__ agg, const float* __restrict__ bias,
                       float* __restrict__ xf, _Float16* __restrict__ xh,
                       int nodes, int Dd, int residual) {
    int t = blockIdx.x * blockDim.x + threadIdx.x;
    if (t >= nodes * Dd) return;
    int c = t % Dd;
    float v = agg[t] + bias[c];
    v = fmaxf(v, 0.f);
    if (residual) v += xf[t];
    xf[t] = v;
    xh[t] = (_Float16)v;
}

// ---------------------------------------------------------------------------
// global mean pool: accumulate sums + counts
// ---------------------------------------------------------------------------
__global__ void k_pool(const int* __restrict__ batch, const float* __restrict__ xf,
                       float* __restrict__ psum, float* __restrict__ cnt,
                       int nodes, int Dd) {
    int t = blockIdx.x * blockDim.x + threadIdx.x;
    if (t >= nodes * Dd) return;
    int n = t / Dd, c = t - n * Dd;
    int g = batch[n];
    atomicAdd(psum + (size_t)g * Dd + c, xf[t]);
    if (c == 0) atomicAdd(cnt + g, 1.f);
}

__global__ void k_pooldiv(const float* __restrict__ psum, const float* __restrict__ cnt,
                          float* __restrict__ pf, _Float16* __restrict__ ph,
                          int G, int Dd) {
    int t = blockIdx.x * blockDim.x + threadIdx.x;
    if (t >= G * Dd) return;
    int g = t / Dd;
    float v = psum[t] / fmaxf(cnt[g], 1.f);
    pf[t] = v;
    ph[t] = (_Float16)v;
}

// ---------------------------------------------------------------------------
// BN(eval) + relu, write f32 and f16
// ---------------------------------------------------------------------------
__global__ void k_bnrelu(const float* __restrict__ in, const float* __restrict__ bias,
                         const float* __restrict__ g, const float* __restrict__ bt,
                         const float* __restrict__ rm, const float* __restrict__ rv,
                         float* __restrict__ outf, _Float16* __restrict__ outh,
                         int M, int Nc) {
    int t = blockIdx.x * blockDim.x + threadIdx.x;
    if (t >= M * Nc) return;
    int c = t % Nc;
    float v = in[t] + bias[c];
    v = (v - rm[c]) * rsqrtf(rv[c] + 1e-5f) * g[c] + bt[c];
    v = fmaxf(v, 0.f);
    outf[t] = v;
    outh[t] = (_Float16)v;
}

// ---------------------------------------------------------------------------
// final projection + sigmoid; one wave per graph
// ---------------------------------------------------------------------------
__global__ void k_final(const float* __restrict__ z, const float* __restrict__ W3,
                        const float* __restrict__ b3, float* __restrict__ out,
                        int G, int Kf) {
    int gt = blockIdx.x * blockDim.x + threadIdx.x;
    int w = gt >> 5, lane = gt & 31;
    if (w >= G) return;
    float s = 0.f;
    for (int c = lane; c < Kf; c += 32) s += z[(size_t)w * Kf + c] * W3[c];
    s = wave_sum32(s);
    if (lane == 0) out[w] = 1.f / (1.f + __expf(-(s + b3[0])));
}

// ---------------------------------------------------------------------------
// host-side orchestration
// ---------------------------------------------------------------------------
extern "C" void kernel_launch(void* const* d_in, const int* in_sizes, int n_in,
                              void* d_out, int out_size, void* d_ws, size_t ws_size,
                              hipStream_t stream) {
    const int Nn   = in_sizes[0];          // 40000 nodes
    const int E    = in_sizes[1] / 2;      // 150000 edges
    const int Etot = E + Nn;               // + self loops
    const int G    = 128;
    const int Dd   = 512;
    const int Mpad = ((Nn + BM - 1) / BM) * BM;

    const int*   node_ids = (const int*)d_in[0];
    const int*   ei       = (const int*)d_in[1];
    const int*   batch    = (const int*)d_in[2];
    const float* emb      = (const float*)d_in[3];
    const float* Wl[3]  = {(const float*)d_in[4], (const float*)d_in[8],  (const float*)d_in[12]};
    const float* asl[3] = {(const float*)d_in[5], (const float*)d_in[9],  (const float*)d_in[13]};
    const float* adl[3] = {(const float*)d_in[6], (const float*)d_in[10], (const float*)d_in[14]};
    const float* bl[3]  = {(const float*)d_in[7], (const float*)d_in[11], (const float*)d_in[15]};

    // workspace bump allocator
    char* wp = (char*)d_ws;
    auto alloc = [&](size_t bytes) -> void* {
        void* p = (void*)wp;
        wp += (bytes + 255) & ~(size_t)255;
        return p;
    };
    _Float16* xh   = (_Float16*)alloc((size_t)Mpad * Dd * 2);
    float*    xf   = (float*)alloc((size_t)Mpad * Dd * 4);
    float*    hf   = (float*)alloc((size_t)Mpad * Dd * 4);
    float*    agg  = (float*)alloc((size_t)Nn * Dd * 4);
    float*    es   = (float*)alloc((size_t)Nn * 4 * 4);
    float*    ed   = (float*)alloc((size_t)Nn * 4 * 4);
    unsigned* maxd = (unsigned*)alloc((size_t)Nn * 4 * 4);
    float*    sumd = (float*)alloc((size_t)Nn * 4 * 4);
    float*    pe   = (float*)alloc((size_t)Etot * 4 * 4);
    _Float16* Wh[3];
    for (int l = 0; l < 3; ++l) Wh[l] = (_Float16*)alloc((size_t)Dd * Dd * 2);
    _Float16* hW1[2] = {(_Float16*)alloc((size_t)Dd * Dd * 2), (_Float16*)alloc((size_t)Dd * Dd * 2)};
    _Float16* hW2[2] = {(_Float16*)alloc((size_t)Dd * (Dd/2) * 2), (_Float16*)alloc((size_t)Dd * (Dd/2) * 2)};
    float*    psum = (float*)alloc((size_t)G * Dd * 4);
    float*    cnt  = (float*)alloc((size_t)G * 4);
    float*    pf   = (float*)alloc((size_t)G * Dd * 4);
    _Float16* ph   = (_Float16*)alloc((size_t)G * Dd * 2);
    float*    g1   = (float*)alloc((size_t)G * Dd * 4);
    float*    z1f  = (float*)alloc((size_t)G * Dd * 4);
    _Float16* z1h  = (_Float16*)alloc((size_t)G * Dd * 2);
    float*    g2   = (float*)alloc((size_t)G * (Dd/2) * 4);
    float*    z2f  = (float*)alloc((size_t)G * (Dd/2) * 4);
    _Float16* z2h  = (_Float16*)alloc((size_t)G * (Dd/2) * 2);

    const int TB = 256;
    auto blocks = [&](long long n) { return (unsigned)((n + TB - 1) / TB); };

    // cast weights to f16
    for (int l = 0; l < 3; ++l)
        k_cast_h<<<blocks(Dd * Dd), TB, 0, stream>>>(Wl[l], Wh[l], Dd * Dd);
    k_cast_h<<<blocks(Dd * Dd),     TB, 0, stream>>>((const float*)d_in[16], hW1[0], Dd * Dd);
    k_cast_h<<<blocks(Dd * Dd),     TB, 0, stream>>>((const float*)d_in[30], hW1[1], Dd * Dd);
    k_cast_h<<<blocks(Dd * Dd / 2), TB, 0, stream>>>((const float*)d_in[22], hW2[0], Dd * Dd / 2);
    k_cast_h<<<blocks(Dd * Dd / 2), TB, 0, stream>>>((const float*)d_in[36], hW2[1], Dd * Dd / 2);

    // zero padded tail rows of xh so WMMA tiles are defined
    hipMemsetAsync(xh + (size_t)Nn * Dd, 0, (size_t)(Mpad - Nn) * Dd * 2, stream);

    // embedding
    k_embed<<<blocks((long long)Nn * Dd), TB, 0, stream>>>(node_ids, emb, xf, xh, Nn, Dd);

    // GAT layers
    dim3 ggrid(Mpad / BM, Dd / BN);
    for (int l = 0; l < 3; ++l) {
        const int heads = (l < 2) ? 4 : 1;
        const int ch    = Dd / heads;
        k_gemm_wmma<<<ggrid, TB, 0, stream>>>(xh, Wh[l], hf, Mpad, Dd, Dd);
        k_scores<<<blocks((long long)Nn * heads * 32), TB, 0, stream>>>(
            hf, asl[l], adl[l], es, ed, Nn, heads, ch);
        hipMemsetAsync(maxd, 0, (size_t)Nn * heads * 4, stream);   // enc^-1(0) < all reals
        k_edge_max<<<blocks((long long)Etot * heads), TB, 0, stream>>>(
            ei, es, ed, maxd, E, Etot, heads);
        hipMemsetAsync(sumd, 0, (size_t)Nn * heads * 4, stream);
        k_edge_p<<<blocks((long long)Etot * heads), TB, 0, stream>>>(
            ei, es, ed, maxd, pe, sumd, E, Etot, heads);
        hipMemsetAsync(agg, 0, (size_t)Nn * Dd * 4, stream);
        k_aggregate<<<blocks((long long)Etot * 32), TB, 0, stream>>>(
            ei, hf, pe, sumd, agg, E, Etot, heads, ch);
        k_post<<<blocks((long long)Nn * Dd), TB, 0, stream>>>(
            agg, bl[l], xf, xh, Nn, Dd, l > 0 ? 1 : 0);
    }

    // global mean pool
    hipMemsetAsync(psum, 0, (size_t)G * Dd * 4, stream);
    hipMemsetAsync(cnt, 0, (size_t)G * 4, stream);
    k_pool<<<blocks((long long)Nn * Dd), TB, 0, stream>>>(batch, xf, psum, cnt, Nn, Dd);
    k_pooldiv<<<blocks((long long)G * Dd), TB, 0, stream>>>(psum, cnt, pf, ph, G, Dd);

    // heads: 0 = danger (inputs 16..29, out offset 0); 1 = synergy (30..43, offset G)
    for (int hsel = 0; hsel < 2; ++hsel) {
        const int b = (hsel == 0) ? 16 : 30;
        const float* hb1  = (const float*)d_in[b + 1];
        const float* hg1  = (const float*)d_in[b + 2];
        const float* hbt1 = (const float*)d_in[b + 3];
        const float* hrm1 = (const float*)d_in[b + 4];
        const float* hrv1 = (const float*)d_in[b + 5];
        const float* hb2  = (const float*)d_in[b + 7];
        const float* hg2  = (const float*)d_in[b + 8];
        const float* hbt2 = (const float*)d_in[b + 9];
        const float* hrm2 = (const float*)d_in[b + 10];
        const float* hrv2 = (const float*)d_in[b + 11];
        const float* hW3  = (const float*)d_in[b + 12];
        const float* hb3  = (const float*)d_in[b + 13];

        dim3 hg1grid(G / BM, Dd / BN);
        k_gemm_wmma<<<hg1grid, TB, 0, stream>>>(ph, hW1[hsel], g1, G, Dd, Dd);
        k_bnrelu<<<blocks((long long)G * Dd), TB, 0, stream>>>(
            g1, hb1, hg1, hbt1, hrm1, hrv1, z1f, z1h, G, Dd);
        dim3 hg2grid(G / BM, (Dd / 2) / BN);
        k_gemm_wmma<<<hg2grid, TB, 0, stream>>>(z1h, hW2[hsel], g2, G, Dd / 2, Dd);
        k_bnrelu<<<blocks((long long)G * Dd / 2), TB, 0, stream>>>(
            g2, hb2, hg2, hbt2, hrm2, hrv2, z2f, z2h, G, Dd / 2);
        k_final<<<blocks((long long)G * 32), TB, 0, stream>>>(
            z2f, hW3, hb3, (float*)d_out + hsel * G, G, Dd / 2);
    }
    (void)n_in; (void)out_size; (void)ws_size;
}
